// GINEncoder_56427280335346
// MI455X (gfx1250) — matrix operand order repaired and checked by reading, not compiled, not run on hardware
//
#include <hip/hip_runtime.h>
#include <hip/hip_bf16.h>

// ---------------------------------------------------------------------------
// GIN encoder for MI455X (gfx1250): bf16 WMMA GEMMs (v_wmma_f32_16x16x32_bf16),
// f32 elementwise/atomics (float4-vectorized), fused 2-stage node MLP via LDS.
// ---------------------------------------------------------------------------

typedef __bf16 bf16;
typedef __attribute__((ext_vector_type(16))) __bf16 v16bf;
typedef __attribute__((ext_vector_type(8)))  __bf16 v8bf;
typedef __attribute__((ext_vector_type(8)))  float  v8f;

static constexpr int N_  = 100000;
static constexpr int E_  = 600000;
static constexpr int G_  = 2048;
static constexpr int DIN_= 9;
static constexpr int DD  = 128;
static constexpr int L_  = 5;

__device__ __forceinline__ bf16 f2bf(float f) { return (bf16)f; }

// A-fragment (16x32 bf16, MxK) from an LDS tile of bf16 rows.
// ISA layout: lane l -> row l&15; element e -> K = k0 + (l>>4)*8 + (e<8 ? e : 16+e-8)
__device__ __forceinline__ v16bf load_a_frag(const bf16* tile, int rowStride,
                                             int mBase, int k0, int lane) {
  int row  = mBase + (lane & 15);
  int koff = k0 + ((lane >> 4) << 3);
  const bf16* p = tile + row * rowStride + koff;
  v8bf lo = *(const v8bf*)(p);
  v8bf hi = *(const v8bf*)(p + 16);
  v16bf a;
#pragma unroll
  for (int e = 0; e < 8; ++e) { a[e] = lo[e]; a[e + 8] = hi[e]; }
  return a;
}

// B-fragment (32x16 bf16, KxN) from pre-packed global weights:
// fragment = 32 lanes x 16 contiguous bf16 (lane l -> col l&15, K = (l>>4)*8 + (e<8 ? e : 16+e-8))
__device__ __forceinline__ v16bf load_b_frag(const bf16* wp, int fragIdx, int lane) {
  return *(const v16bf*)(wp + fragIdx * 512 + lane * 16);
}

// --------------------------- small utility kernels -------------------------

__global__ void zero4_kernel(float4* __restrict__ p, int n4) {
  int i = blockIdx.x * blockDim.x + threadIdx.x;
  if (i < n4) p[i] = make_float4(0.f, 0.f, 0.f, 0.f);
}

__global__ void copy4_kernel(const float4* __restrict__ a, float4* __restrict__ b, int n4) {
  int i = blockIdx.x * blockDim.x + threadIdx.x;
  if (i < n4) b[i] = a[i];
}

__global__ void count_kernel(const int* __restrict__ batch, float* __restrict__ cnt, int n) {
  int i = blockIdx.x * blockDim.x + threadIdx.x;
  if (i < n) atomicAdd(&cnt[batch[i]], 1.0f);
}

__global__ void vn_init_kernel(const float4* __restrict__ emb, float4* __restrict__ vn) {
  int i = blockIdx.x * blockDim.x + threadIdx.x;
  if (i < G_ * (DD / 4)) vn[i] = emb[i & (DD / 4 - 1)];
}

// h[r, f0..f0+3] = x[r,:] @ W[:, f0..f0+3] + b ; one thread per 4 outputs
__global__ void node_embed_kernel(const float* __restrict__ x, const float* __restrict__ W,
                                  const float* __restrict__ b, float* __restrict__ h) {
  int i = blockIdx.x * blockDim.x + threadIdx.x;
  if (i >= N_ * (DD / 4)) return;
  int r = i >> 5, f0 = (i & 31) * 4;
  float4 acc = *(const float4*)(b + f0);
#pragma unroll
  for (int k = 0; k < DIN_; ++k) {
    float xv = x[r * DIN_ + k];
    float4 wv = *(const float4*)(W + k * DD + f0);
    acc.x += xv * wv.x; acc.y += xv * wv.y; acc.z += xv * wv.z; acc.w += xv * wv.w;
  }
  *(float4*)(h + r * DD + f0) = acc;
}

// m[dst] += h[src] per edge; 32 threads per edge, float4 gather + 4 atomics
__global__ void scatter_kernel(const float* __restrict__ h, const int* __restrict__ ei,
                               float* __restrict__ m) {
  int i = blockIdx.x * blockDim.x + threadIdx.x;
  if (i >= E_ * 32) return;
  int e = i >> 5, f0 = (i & 31) * 4;
  int src = ei[e], dst = ei[E_ + e];
  float4 v = *(const float4*)(h + src * DD + f0);
  float* p = m + dst * DD + f0;
  atomicAdd(p + 0, v.x);
  atomicAdd(p + 1, v.y);
  atomicAdd(p + 2, v.z);
  atomicAdd(p + 3, v.w);
}

// per-feature sum / sumsq over rows (coalesced 512B row reads)
__global__ void bn_stats_kernel(const float* __restrict__ y, float* __restrict__ stats, int n) {
  int f = threadIdx.x;               // blockDim = 128
  float s = 0.f, ss = 0.f;
  for (int r = blockIdx.x; r < n; r += gridDim.x) {
    float v = y[r * DD + f];
    s += v; ss += v * v;
  }
  atomicAdd(&stats[f], s);
  atomicAdd(&stats[DD + f], ss);
}

// h = relu(BN(y)) (+h residual) ; also accumulate mean-pool numerator
__global__ void bn_apply_kernel(const float* __restrict__ y, float* __restrict__ h,
                                const float* __restrict__ stats,
                                const float* __restrict__ g, const float* __restrict__ b,
                                const int* __restrict__ batch, float* __restrict__ pool,
                                int addRes) {
  int i = blockIdx.x * blockDim.x + threadIdx.x;
  if (i >= N_ * (DD / 4)) return;
  int r = i >> 5, f0 = (i & 31) * 4;
  const float inv_n = 1.0f / (float)N_;
  int gidx = batch[r];
  float4 yv4 = *(const float4*)(y + r * DD + f0);
  float4 hv4 = *(const float4*)(h + r * DD + f0);
  float4 sm4 = *(const float4*)(stats + f0);
  float4 sq4 = *(const float4*)(stats + DD + f0);
  float4 gg4 = *(const float4*)(g + f0);
  float4 bb4 = *(const float4*)(b + f0);
  const float* yv = &yv4.x; const float* hv = &hv4.x;
  const float* sm = &sm4.x; const float* sq = &sq4.x;
  const float* gg = &gg4.x; const float* bb = &bb4.x;
  float4 out4; float* out = &out4.x;
#pragma unroll
  for (int j = 0; j < 4; ++j) {
    float mu  = sm[j] * inv_n;
    float var = sq[j] * inv_n - mu * mu;
    float v = (yv[j] - mu) * rsqrtf(var + 1e-5f) * gg[j] + bb[j];
    v = fmaxf(v, 0.f);
    if (addRes) v += hv[j];
    out[j] = v;
  }
  *(float4*)(h + r * DD + f0) = out4;
  float* pp = pool + gidx * DD + f0;
  atomicAdd(pp + 0, out[0]);
  atomicAdd(pp + 1, out[1]);
  atomicAdd(pp + 2, out[2]);
  atomicAdd(pp + 3, out[3]);
}

__global__ void vn_bcast_kernel(float* __restrict__ h, const float* __restrict__ vn,
                                const int* __restrict__ batch) {
  int i = blockIdx.x * blockDim.x + threadIdx.x;
  if (i >= N_ * (DD / 4)) return;
  int r = i >> 5, f0 = (i & 31) * 4;
  float4 vv = *(const float4*)(vn + batch[r] * DD + f0);
  float4 hv = *(const float4*)(h + r * DD + f0);
  hv.x += vv.x; hv.y += vv.y; hv.z += vv.z; hv.w += vv.w;
  *(float4*)(h + r * DD + f0) = hv;
}

__global__ void pool_acc_kernel(const float* __restrict__ h, const int* __restrict__ batch,
                                float* __restrict__ pool) {
  int i = blockIdx.x * blockDim.x + threadIdx.x;
  if (i >= N_ * (DD / 4)) return;
  int r = i >> 5, f0 = (i & 31) * 4;
  float4 v = *(const float4*)(h + r * DD + f0);
  float* pp = pool + batch[r] * DD + f0;
  atomicAdd(pp + 0, v.x);
  atomicAdd(pp + 1, v.y);
  atomicAdd(pp + 2, v.z);
  atomicAdd(pp + 3, v.w);
}

// row-wise LayerNorm + ReLU over 128 features, one wave32 per row (lane owns 4)
__global__ void ln_relu_kernel(float* __restrict__ gf, const float* __restrict__ g,
                               const float* __restrict__ b) {
  int row  = blockIdx.x * (blockDim.x >> 5) + (threadIdx.x >> 5);
  int lane = threadIdx.x & 31;
  if (row >= G_) return;
  int f0 = lane * 4;
  float4 v4 = *(const float4*)(gf + row * DD + f0);
  const float* v = &v4.x;
  float s = v[0] + v[1] + v[2] + v[3];
#pragma unroll
  for (int off = 16; off; off >>= 1) s += __shfl_xor(s, off, 32);
  float mu = s * (1.0f / DD);
  float ss = 0.f;
#pragma unroll
  for (int j = 0; j < 4; ++j) { float d = v[j] - mu; ss += d * d; }
#pragma unroll
  for (int off = 16; off; off >>= 1) ss += __shfl_xor(ss, off, 32);
  float rstd = rsqrtf(ss * (1.0f / DD) + 1e-5f);
  float4 g4 = *(const float4*)(g + f0);
  float4 b4 = *(const float4*)(b + f0);
  const float* gg = &g4.x; const float* bb = &b4.x;
  float4 o4; float* o = &o4.x;
#pragma unroll
  for (int j = 0; j < 4; ++j)
    o[j] = fmaxf((v[j] - mu) * rstd * gg[j] + bb[j], 0.f);
  *(float4*)(gf + row * DD + f0) = o4;
}

// pack fp32 weight [Ktot x Ntot] row-major into fragment-native bf16 layout
__global__ void pack_w_kernel(const float* __restrict__ W, bf16* __restrict__ out,
                              int Ktot, int Ntot) {
  int p = blockIdx.x * blockDim.x + threadIdx.x;
  if (p >= Ktot * Ntot) return;
  int frag = p >> 9, r = p & 511;
  int lane = r >> 4, e = r & 15;
  int ntiles = Ntot >> 4;
  int nt = frag % ntiles, kt = frag / ntiles;
  int k = kt * 32 + ((lane >> 4) << 3) + (e < 8 ? e : 16 + e - 8);
  int n = nt * 16 + (lane & 15);
  out[p] = f2bf(W[k * Ntot + n]);
}

// ------------------ fused node MLP: relu(m@W1+b1)@W2+b2, in-place -----------

__global__ __launch_bounds__(128) void gin_mlp_kernel(
    const float* __restrict__ m_in, float* __restrict__ out,
    const bf16* __restrict__ w1p, const float* __restrict__ b1,
    const bf16* __restrict__ w2p, const float* __restrict__ b2, int n) {
  __shared__ __attribute__((aligned(32))) bf16 aLds[64 * 128];
  __shared__ __attribute__((aligned(32))) bf16 tLds[64 * 256];
  int tid = threadIdx.x, lane = tid & 31, wave = tid >> 5;
  int base = blockIdx.x * 64;
  for (int i = tid; i < 64 * 128; i += 128) {
    int r = i >> 7, c = i & 127;
    int gr = base + r; if (gr >= n) gr = n - 1;
    aLds[i] = f2bf(m_in[gr * DD + c]);
  }
  __syncthreads();
  int mBase = wave * 16;
  int rowLo = (lane >> 4) << 3;
  int colL  = lane & 15;
  // stage 1: T = relu(A @ W1 + b1), N=256, K=128
  for (int nt = 0; nt < 16; ++nt) {
    v8f acc = {};
#pragma unroll
    for (int kt = 0; kt < 4; ++kt) {
      v16bf a = load_a_frag(aLds, 128, mBase, kt * 32, lane);
      v16bf w = load_b_frag(w1p, kt * 16 + nt, lane);
      acc = __builtin_amdgcn_wmma_f32_16x16x32_bf16(false, a, false, w, (short)0, acc, false, false);
    }
    int col = nt * 16 + colL;
    float bias = b1[col];
    int row0 = mBase + rowLo;
#pragma unroll
    for (int j = 0; j < 8; ++j)
      tLds[(row0 + j) * 256 + col] = f2bf(fmaxf(acc[j] + bias, 0.f));
  }
  __syncthreads();
  // stage 2: Y = T @ W2 + b2, N=128, K=256
  for (int nt = 0; nt < 8; ++nt) {
    v8f acc = {};
#pragma unroll
    for (int kt = 0; kt < 8; ++kt) {
      v16bf a = load_a_frag(tLds, 256, mBase, kt * 32, lane);
      v16bf w = load_b_frag(w2p, kt * 8 + nt, lane);
      acc = __builtin_amdgcn_wmma_f32_16x16x32_bf16(false, a, false, w, (short)0, acc, false, false);
    }
    int col = nt * 16 + colL;
    float bias = b2[col];
    int row0 = base + mBase + rowLo;
#pragma unroll
    for (int j = 0; j < 8; ++j)
      if (row0 + j < n) out[(row0 + j) * DD + col] = acc[j] + bias;
  }
}

// -------- generic [M,128]@[128,128] WMMA GEMM with prep/epilogue flags ------
// flags: 1 = divide input row by max(cnt,1) (mean-pool prep)
//        2 = relu epilogue
//        4 = accumulate into output
__global__ __launch_bounds__(128) void gemm128_kernel(
    const float* __restrict__ A, const bf16* __restrict__ wp,
    const float* __restrict__ bias, float* __restrict__ outp,
    const float* __restrict__ cnt, int M, int flags) {
  __shared__ __attribute__((aligned(32))) bf16 aLds[64 * 128];
  int tid = threadIdx.x, lane = tid & 31, wave = tid >> 5;
  int base = blockIdx.x * 64;
  for (int i = tid; i < 64 * 128; i += 128) {
    int r = i >> 7, c = i & 127;
    int gr = base + r; if (gr >= M) gr = M - 1;
    float v = A[gr * DD + c];
    if (flags & 1) v /= fmaxf(cnt[gr], 1.0f);
    aLds[i] = f2bf(v);
  }
  __syncthreads();
  int mBase = wave * 16;
  int rowLo = (lane >> 4) << 3;
  int colL  = lane & 15;
  for (int nt = 0; nt < 8; ++nt) {
    v8f acc = {};
#pragma unroll
    for (int kt = 0; kt < 4; ++kt) {
      v16bf a = load_a_frag(aLds, 128, mBase, kt * 32, lane);
      v16bf w = load_b_frag(wp, kt * 8 + nt, lane);
      acc = __builtin_amdgcn_wmma_f32_16x16x32_bf16(false, a, false, w, (short)0, acc, false, false);
    }
    int col = nt * 16 + colL;
    float bb = bias[col];
    int row0 = base + mBase + rowLo;
#pragma unroll
    for (int j = 0; j < 8; ++j) {
      if (row0 + j < M) {
        float v = acc[j] + bb;
        if (flags & 2) v = fmaxf(v, 0.f);
        if (flags & 4) outp[(row0 + j) * DD + col] += v;
        else           outp[(row0 + j) * DD + col]  = v;
      }
    }
  }
}

// ------------------------------- host side ----------------------------------

static inline int cdiv(int a, int b) { return (a + b - 1) / b; }

extern "C" void kernel_launch(void* const* d_in, const int* in_sizes, int n_in,
                              void* d_out, int out_size, void* d_ws, size_t ws_size,
                              hipStream_t stream) {
  const float* x        = (const float*)d_in[0];
  const int*   ei       = (const int*)  d_in[1];
  const int*   batch    = (const int*)  d_in[2];
  const float* emb_W    = (const float*)d_in[3];
  const float* emb_b    = (const float*)d_in[4];
  const float* W1s      = (const float*)d_in[5];
  const float* b1s      = (const float*)d_in[6];
  const float* W2s      = (const float*)d_in[7];
  const float* b2s      = (const float*)d_in[8];
  const float* bn_g     = (const float*)d_in[9];
  const float* bn_b     = (const float*)d_in[10];
  const float* vn_emb   = (const float*)d_in[11];
  const float* vn_W1    = (const float*)d_in[12];
  const float* vn_b1    = (const float*)d_in[13];
  const float* vn_W2    = (const float*)d_in[14];
  const float* vn_b2    = (const float*)d_in[15];
  const float* gm_W1    = (const float*)d_in[16];
  const float* gm_b1    = (const float*)d_in[17];
  const float* ln_g     = (const float*)d_in[18];
  const float* ln_b     = (const float*)d_in[19];
  const float* gm_W2    = (const float*)d_in[20];
  const float* gm_b2    = (const float*)d_in[21];

  float* ws = (float*)d_ws;
  size_t off = 0;
  float* h     = ws + off; off += (size_t)N_ * DD;
  float* m     = ws + off; off += (size_t)N_ * DD;
  float* pool  = ws + off; off += (size_t)G_ * DD;
  float* vn    = ws + off; off += (size_t)G_ * DD;
  float* gtmp  = ws + off; off += (size_t)G_ * DD;
  float* cnt   = ws + off; off += G_;
  float* stats = ws + off; off += 256;
  off = (off + 63) & ~(size_t)63;                   // 32B-align bf16 region
  bf16* bfbase = (bf16*)(ws + off);
  bf16* W1p   = bfbase;                 // L * 128*256
  bf16* W2p   = W1p  + (size_t)L_ * DD * 2 * DD;
  bf16* vnW1p = W2p  + (size_t)L_ * 2 * DD * DD;
  bf16* vnW2p = vnW1p + DD * DD;
  bf16* gmW1p = vnW2p + DD * DD;
  bf16* gmW2p = gmW1p + DD * DD;

  const int TB  = 256;
  const int ND  = N_ * DD, GD = G_ * DD;
  const int NQ  = N_ * (DD / 4);          // float4 work-items per node pass
  const int EQ  = E_ * 32;                // float4 work-items per edge pass

  // ---- prep ----
  zero4_kernel<<<cdiv(G_ / 4, TB), TB, 0, stream>>>((float4*)cnt, G_ / 4);
  count_kernel<<<cdiv(N_, TB), TB, 0, stream>>>(batch, cnt, N_);
  node_embed_kernel<<<cdiv(NQ, TB), TB, 0, stream>>>(x, emb_W, emb_b, h);
  vn_init_kernel<<<cdiv(GD / 4, TB), TB, 0, stream>>>((const float4*)vn_emb, (float4*)vn);
  for (int i = 0; i < L_; ++i) {
    pack_w_kernel<<<cdiv(DD * 2 * DD, TB), TB, 0, stream>>>(W1s + (size_t)i * DD * 2 * DD,
                                                            W1p + (size_t)i * DD * 2 * DD, DD, 2 * DD);
    pack_w_kernel<<<cdiv(2 * DD * DD, TB), TB, 0, stream>>>(W2s + (size_t)i * 2 * DD * DD,
                                                            W2p + (size_t)i * 2 * DD * DD, 2 * DD, DD);
  }
  pack_w_kernel<<<cdiv(DD * DD, TB), TB, 0, stream>>>(vn_W1, vnW1p, DD, DD);
  pack_w_kernel<<<cdiv(DD * DD, TB), TB, 0, stream>>>(vn_W2, vnW2p, DD, DD);
  pack_w_kernel<<<cdiv(DD * DD, TB), TB, 0, stream>>>(gm_W1, gmW1p, DD, DD);
  pack_w_kernel<<<cdiv(DD * DD, TB), TB, 0, stream>>>(gm_W2, gmW2p, DD, DD);

  // ---- GIN layers ----
  for (int i = 0; i < L_; ++i) {
    copy4_kernel<<<cdiv(ND / 4, TB), TB, 0, stream>>>((const float4*)h, (float4*)m, ND / 4);
    scatter_kernel<<<cdiv(EQ, TB), TB, 0, stream>>>(h, ei, m);     // m = h + sum_{j->i} h_j
    gin_mlp_kernel<<<cdiv(N_, 64), 128, 0, stream>>>(m, m,
        W1p + (size_t)i * DD * 2 * DD, b1s + (size_t)i * 2 * DD,
        W2p + (size_t)i * 2 * DD * DD, b2s + (size_t)i * DD, N_);
    zero4_kernel<<<1, 64, 0, stream>>>((float4*)stats, 64);
    bn_stats_kernel<<<256, DD, 0, stream>>>(m, stats, N_);
    zero4_kernel<<<cdiv(GD / 4, TB), TB, 0, stream>>>((float4*)pool, GD / 4);
    bn_apply_kernel<<<cdiv(NQ, TB), TB, 0, stream>>>(m, h, stats,
        bn_g + (size_t)i * DD, bn_b + (size_t)i * DD, batch, pool, i > 0 ? 1 : 0);
    // virtual node: vn += relu(mean_pool @ vnW1 + b1) @ vnW2 + b2
    gemm128_kernel<<<G_ / 64, 128, 0, stream>>>(pool, vnW1p, vn_b1, gtmp, cnt, G_, 1 | 2);
    gemm128_kernel<<<G_ / 64, 128, 0, stream>>>(gtmp, vnW2p, vn_b2, vn,  cnt, G_, 4);
    vn_bcast_kernel<<<cdiv(NQ, TB), TB, 0, stream>>>(h, vn, batch);
  }

  // ---- outputs ----
  hipMemcpyAsync(d_out, h, (size_t)ND * sizeof(float), hipMemcpyDeviceToDevice, stream);
  zero4_kernel<<<cdiv(GD / 4, TB), TB, 0, stream>>>((float4*)pool, GD / 4);
  pool_acc_kernel<<<cdiv(NQ, TB), TB, 0, stream>>>(h, batch, pool);
  gemm128_kernel<<<G_ / 64, 128, 0, stream>>>(pool, gmW1p, gm_b1, gtmp, cnt, G_, 1);
  ln_relu_kernel<<<G_ / 8, 256, 0, stream>>>(gtmp, ln_g, ln_b);
  gemm128_kernel<<<G_ / 64, 128, 0, stream>>>(gtmp, gmW2p, gm_b2,
                                              (float*)d_out + (size_t)ND, cnt, G_, 0);
}
